// RNNModel_16870631539166
// MI455X (gfx1250) — matrix operand order repaired and checked by reading, not compiled, not run on hardware
//
#include <hip/hip_runtime.h>

#define BATCH 2048
#define TSTEPS 512
#define ISZ 5
#define HSZ 64
#define HSTR 72      // padded LDS row stride for h tiles (f16 elems, 144B)
#define CT 32        // timesteps per streamed x chunk
#define XROW_DW 164  // padded x-buffer row stride in dwords (160 data + 4 pad)
#define NCHUNK (TSTEPS / CT)

typedef _Float16 v16h __attribute__((ext_vector_type(16)));
typedef float    v8f  __attribute__((ext_vector_type(8)));
typedef float    f32x4 __attribute__((ext_vector_type(4)));

__device__ __forceinline__ float tanh_hw(float x) {
#if __has_builtin(__builtin_amdgcn_tanhf)
    return __builtin_amdgcn_tanhf(x);
#else
    float y;
    asm volatile("v_tanh_f32 %0, %1\n\tv_nop" : "=v"(y) : "v"(x));
    return y;
#endif
}

__device__ __forceinline__ void tanh8(v8f& c) {
#pragma unroll
    for (int r = 0; r < 8; ++r) c[r] = tanh_hw(c[r]);
}

__device__ __forceinline__ v8f wmma_f16(v16h a, v16h b, v8f c) {
    return __builtin_amdgcn_wmma_f32_16x16x32_f16(false, a, false, b, (short)0, c,
                                                  false, false);
}

// ---- gfx1250 async copy: global -> LDS, 16B per lane, tracked by ASYNCcnt ----
__device__ __forceinline__ void async_ld_b128(unsigned int lds_off, const void* g) {
    asm volatile("global_load_async_to_lds_b128 %0, %1, off"
                 :: "v"(lds_off), "v"((unsigned long long)(uintptr_t)g)
                 : "memory");
}
__device__ __forceinline__ void wait_async_5() {
    asm volatile("s_wait_asynccnt 5" ::: "memory");
}
__device__ __forceinline__ void wait_async_0() {
    asm volatile("s_wait_asynccnt 0" ::: "memory");
}
__device__ __forceinline__ unsigned int lds_off32(const void* p) {
    return (unsigned int)(uintptr_t)p;
}

// Load 16 consecutive f32 weights, convert to an f16 B-fragment slice.
__device__ __forceinline__ v16h load_w16(const float* __restrict__ p) {
    f32x4 a = *(const f32x4*)(p);
    f32x4 b = *(const f32x4*)(p + 4);
    f32x4 c = *(const f32x4*)(p + 8);
    f32x4 d = *(const f32x4*)(p + 12);
    v16h r;
#pragma unroll
    for (int i = 0; i < 4; ++i) {
        r[i]      = (_Float16)a[i];
        r[4 + i]  = (_Float16)b[i];
        r[8 + i]  = (_Float16)c[i];
        r[12 + i] = (_Float16)d[i];
    }
    return r;
}

// A-fragment (16x32 f16) for k-block kb from an LDS hidden-state tile.
__device__ __forceinline__ v16h load_a_frag(const _Float16* hs, int m, int kb, int lhalf) {
    const _Float16* base = hs + m * HSTR + kb * 32 + lhalf * 16;
    union { v16h v; f32x4 f[2]; } u;
    u.f[0] = *(const f32x4*)(base);
    u.f[1] = *(const f32x4*)(base + 8);
    return u.v;
}

// Scatter a C tile (v8f) into LDS as f16: row = r + 8*lhalf, col = wave*16 + l15.
__device__ __forceinline__ void store_h(_Float16* hs, const v8f& c, int l15, int lhalf,
                                        int wave) {
    _Float16* p = hs + (lhalf * 8) * HSTR + wave * 16 + l15;
#pragma unroll
    for (int r = 0; r < 8; ++r) p[r * HSTR] = (_Float16)c[r];
}

// Stream one 32-step x chunk (16 rows x 640B) into LDS: 640 b128 transfers,
// exactly 5 per thread -> uniform per-wave ASYNCcnt increment of 5.
__device__ __forceinline__ void issue_x_chunk(const float* __restrict__ x, int bbase,
                                              int c, float* ldsx, int tid) {
    const int tc = c * CT;
#pragma unroll
    for (int s = 0; s < 5; ++s) {
        int j   = tid + 128 * s;   // 0..639
        int row = j / 40;
        int k   = j - row * 40;
        const float* g = x + ((size_t)(bbase + row) * TSTEPS + tc) * ISZ + k * 4;
        async_ld_b128(lds_off32(ldsx + row * XROW_DW + k * 4), g);
    }
}

__device__ __forceinline__ v16h load_x_frag(const float* xbuf, int trel, int l15,
                                            int lhalf) {
    v16h ax = {};
    if (lhalf == 0) {
        const float* xp = xbuf + l15 * XROW_DW + trel * ISZ;
        ax[0] = (_Float16)xp[0]; ax[1] = (_Float16)xp[1];
        ax[2] = (_Float16)xp[2]; ax[3] = (_Float16)xp[3];
        ax[4] = (_Float16)xp[4];
    }
    return ax;
}

// Skewed step u: computes h1(u) AND h2(u-1). Reads only previous-iteration
// state -> two independent WMMA chains, one barrier, shared h1 A-fragments.
__device__ __forceinline__ void fused_step(
    int trel, const float* xbuf,
    const _Float16* h1r, _Float16* h1w, const _Float16* h2r, _Float16* h2w,
    const v16h& Bih0f, const v16h* Bhh0f, const v16h* Bih1f, const v16h* Bhh1f,
    float bias1, float bias2, int l15, int lhalf, int wave)
{
    v16h a1k0 = load_a_frag(h1r, l15, 0, lhalf);   // h1(u-1)
    v16h a1k1 = load_a_frag(h1r, l15, 1, lhalf);
    v16h a2k0 = load_a_frag(h2r, l15, 0, lhalf);   // h2(u-2)
    v16h a2k1 = load_a_frag(h2r, l15, 1, lhalf);
    v16h ax   = load_x_frag(xbuf, trel, l15, lhalf);

    v8f c1 = {bias1, bias1, bias1, bias1, bias1, bias1, bias1, bias1};
    v8f c2 = {bias2, bias2, bias2, bias2, bias2, bias2, bias2, bias2};
    // interleave the two independent accumulation chains
    c1 = wmma_f16(ax,   Bih0f,    c1);
    c2 = wmma_f16(a1k0, Bih1f[0], c2);
    c1 = wmma_f16(a1k0, Bhh0f[0], c1);
    c2 = wmma_f16(a1k1, Bih1f[1], c2);
    c1 = wmma_f16(a1k1, Bhh0f[1], c1);
    c2 = wmma_f16(a2k0, Bhh1f[0], c2);
    c2 = wmma_f16(a2k1, Bhh1f[1], c2);
    tanh8(c1);
    tanh8(c2);
    store_h(h1w, c1, l15, lhalf, wave);            // h1(u)
    store_h(h2w, c2, l15, lhalf, wave);            // h2(u-1)
    __syncthreads();
}

__global__ __launch_bounds__(128) void rnn2_fused_kernel(
    const float* __restrict__ x,
    const float* __restrict__ Wih0, const float* __restrict__ Whh0,
    const float* __restrict__ bih0, const float* __restrict__ bhh0,
    const float* __restrict__ Wih1, const float* __restrict__ Whh1,
    const float* __restrict__ bih1, const float* __restrict__ bhh1,
    const float* __restrict__ fcW,  const float* __restrict__ fcb,
    float* __restrict__ out)
{
    __shared__ _Float16 h1s0[16 * HSTR], h1s1[16 * HSTR];
    __shared__ _Float16 h2s0[16 * HSTR], h2s1[16 * HSTR];
    __shared__ float    xb0[16 * XROW_DW], xb1[16 * XROW_DW];

    const int tid   = threadIdx.x;
    const int lane  = tid & 31;
    const int wave  = tid >> 5;
    const int lhalf = lane >> 4;
    const int l15   = lane & 15;
    const int bbase = blockIdx.x * 16;

    // ---- resident weight B-fragments ----
    const int n    = wave * 16 + l15;
    const int kcol = lhalf * 16;
    v16h Bhh0f[2], Bih1f[2], Bhh1f[2];
#pragma unroll
    for (int kb = 0; kb < 2; ++kb) {
        Bhh0f[kb] = load_w16(Whh0 + n * HSZ + kb * 32 + kcol);
        Bih1f[kb] = load_w16(Wih1 + n * HSZ + kb * 32 + kcol);
        Bhh1f[kb] = load_w16(Whh1 + n * HSZ + kb * 32 + kcol);
    }
    v16h Bih0f = {};
    if (lhalf == 0) {
#pragma unroll
        for (int i = 0; i < ISZ; ++i) Bih0f[i] = (_Float16)Wih0[n * ISZ + i];
    }
    const float bias1 = bih0[n] + bhh0[n];
    const float bias2 = bih1[n] + bhh1[n];

    // h2(-1) = 0 lives in h2s1 (read by u=1); stream chunk 0
    for (int i = tid; i < 16 * HSTR; i += 128) h2s1[i] = (_Float16)0;
    issue_x_chunk(x, bbase, 0, xb0, tid);
    wait_async_0();
    __syncthreads();
    issue_x_chunk(x, bbase, 1, xb1, tid);     // prefetch chunk 1 early

    // u = 0 (peeled): h1(0) = tanh(bias1 + x0 Wih0^T); h1(-1)=0, no state reads
    {
        v16h ax = load_x_frag(xb0, 0, l15, lhalf);
        v8f c1 = {bias1, bias1, bias1, bias1, bias1, bias1, bias1, bias1};
        c1 = wmma_f16(ax, Bih0f, c1);
        tanh8(c1);
        store_h(h1s1, c1, l15, lhalf, wave);
        __syncthreads();
    }

    // odd u: read buf1 -> write buf0; even u: read buf0 -> write buf1
    const float* xcur = xb0;
    for (int p = 0; p < 255; ++p) {
        const int u = 2 * p + 1;                     // 1,3,...,509
        fused_step(u & (CT - 1), xcur, h1s1, h1s0, h2s1, h2s0,
                   Bih0f, Bhh0f, Bih1f, Bhh1f, bias1, bias2, l15, lhalf, wave);
        const int v = u + 1;                         // 2,4,...,510
        if ((v & (CT - 1)) == 0) {                   // chunk boundary (uniform)
            const int c = v >> 5;                    // 1..15, chunk c already issued
            if (c + 1 < NCHUNK) {
                issue_x_chunk(x, bbase, c + 1, (c & 1) ? xb0 : xb1, tid);
                wait_async_5();                      // chunk c is now resident
            } else {
                wait_async_0();
            }
            __syncthreads();
            xcur = (c & 1) ? xb1 : xb0;
        }
        fused_step(v & (CT - 1), xcur, h1s0, h1s1, h2s0, h2s1,
                   Bih0f, Bhh0f, Bih1f, Bhh1f, bias1, bias2, l15, lhalf, wave);
    }
    // u = 511 -> h1(511) into h1s0, h2(510) into h2s0
    fused_step(511 & (CT - 1), xcur, h1s1, h1s0, h2s1, h2s0,
               Bih0f, Bhh0f, Bih1f, Bhh1f, bias1, bias2, l15, lhalf, wave);

    // epilogue half-step: h2(511) = tanh(bias2 + h1(511) Wih1^T + h2(510) Whh1^T)
    {
        v16h a1k0 = load_a_frag(h1s0, l15, 0, lhalf);
        v16h a1k1 = load_a_frag(h1s0, l15, 1, lhalf);
        v16h a2k0 = load_a_frag(h2s0, l15, 0, lhalf);
        v16h a2k1 = load_a_frag(h2s0, l15, 1, lhalf);
        v8f c2 = {bias2, bias2, bias2, bias2, bias2, bias2, bias2, bias2};
        c2 = wmma_f16(a1k0, Bih1f[0], c2);
        c2 = wmma_f16(a1k1, Bih1f[1], c2);
        c2 = wmma_f16(a2k0, Bhh1f[0], c2);
        c2 = wmma_f16(a2k1, Bhh1f[1], c2);
        tanh8(c2);
        store_h(h2s1, c2, l15, lhalf, wave);
        __syncthreads();
    }

    // out = h2(511) @ fc_W^T + fc_b   (O = 1)
    if (tid < 16) {
        float acc = fcb[0];
#pragma unroll 8
        for (int j = 0; j < HSZ; ++j) acc += (float)h2s1[tid * HSTR + j] * fcW[j];
        out[bbase + tid] = acc;
    }
}

extern "C" void kernel_launch(void* const* d_in, const int* in_sizes, int n_in,
                              void* d_out, int out_size, void* d_ws, size_t ws_size,
                              hipStream_t stream) {
    (void)in_sizes; (void)n_in; (void)d_ws; (void)ws_size; (void)out_size;
    const float* x    = (const float*)d_in[0];
    const float* Wih0 = (const float*)d_in[1];
    const float* Whh0 = (const float*)d_in[2];
    const float* bih0 = (const float*)d_in[3];
    const float* bhh0 = (const float*)d_in[4];
    const float* Wih1 = (const float*)d_in[5];
    const float* Whh1 = (const float*)d_in[6];
    const float* bih1 = (const float*)d_in[7];
    const float* bhh1 = (const float*)d_in[8];
    const float* fcW  = (const float*)d_in[9];
    const float* fcb  = (const float*)d_in[10];
    float* out = (float*)d_out;

    dim3 grid(BATCH / 16);
    dim3 block(128);
    rnn2_fused_kernel<<<grid, block, 0, stream>>>(x, Wih0, Whh0, bih0, bhh0,
                                                  Wih1, Whh1, bih1, bhh1,
                                                  fcW, fcb, out);
}
